// SkimRNN_695784702286
// MI455X (gfx1250) — matrix operand order repaired and checked by reading, not compiled, not run on hardware
//
#include <hip/hip_runtime.h>
#include <hip/hip_bf16.h>
#include <math.h>

// ---------------- problem constants ----------------
#define B_  64
#define T_  512
#define E_  512
#define L_  1024
#define S_  256
#define V_  50000
#define NC_ 5
#define TAU_ 0.5f
#define KL_ (E_ + L_)   // 1536
#define KS_ (E_ + S_)   // 768

typedef _Float16 v16h __attribute__((ext_vector_type(16)));
typedef float    v8f  __attribute__((ext_vector_type(8)));

__device__ __forceinline__ float sigm_(float x) { return 1.0f / (1.0f + __expf(-x)); }
__device__ __forceinline__ float selu_(float x) {
    const float l = 1.0507009873554804934193f, a = 1.6732632423543772848170f;
    return x > 0.0f ? l * x : l * a * (__expf(x) - 1.0f);
}

// ---------------- prep kernels ----------------
// Fuse [W_ih | W_hh] into one f16 matrix per cell, row-major over K; combine biases.
__global__ void prep_weights(const float* __restrict__ Wihl, const float* __restrict__ Whhl,
                             const float* __restrict__ bihl, const float* __restrict__ bhhl,
                             const float* __restrict__ Wihs, const float* __restrict__ Whhs,
                             const float* __restrict__ bihs, const float* __restrict__ bhhs,
                             _Float16* __restrict__ wfl, _Float16* __restrict__ wfs,
                             float* __restrict__ bl, float* __restrict__ bs) {
    const size_t NL = (size_t)(4 * L_) * KL_;
    const size_t NS = (size_t)(4 * S_) * KS_;
    const size_t TOT = NL + NS + 4 * L_ + 4 * S_;
    for (size_t idx = (size_t)blockIdx.x * blockDim.x + threadIdx.x; idx < TOT;
         idx += (size_t)gridDim.x * blockDim.x) {
        if (idx < NL) {
            size_t r = idx / KL_, k = idx % KL_;
            float v = (k < E_) ? Wihl[r * E_ + k] : Whhl[r * L_ + (k - E_)];
            wfl[idx] = (_Float16)v;
        } else if (idx < NL + NS) {
            size_t j = idx - NL, r = j / KS_, k = j % KS_;
            float v = (k < E_) ? Wihs[r * E_ + k] : Whhs[r * S_ + (k - E_)];
            wfs[j] = (_Float16)v;
        } else if (idx < NL + NS + 4 * L_) {
            size_t j = idx - NL - NS;
            bl[j] = bihl[j] + bhhl[j];
        } else {
            size_t j = idx - NL - NS - 4 * L_;
            bs[j] = bihs[j] + bhhs[j];
        }
    }
}

// Embedding gather -> f16 [B,T,E]
__global__ void prep_embed(const int* __restrict__ x, const float* __restrict__ emb,
                           _Float16* __restrict__ embh) {
    const size_t N = (size_t)B_ * T_ * E_;
    for (size_t idx = (size_t)blockIdx.x * blockDim.x + threadIdx.x; idx < N;
         idx += (size_t)gridDim.x * blockDim.x) {
        size_t bt = idx >> 9;           // /E_
        int k = (int)(idx & (E_ - 1));
        embh[idx] = (_Float16)emb[(size_t)x[bt] * E_ + k];
    }
}

// Zero parity-0 state buffers
__global__ void prep_zero(float* hlf, _Float16* hlh, float* clf,
                          float* hsf, _Float16* hsh, float* csf) {
    const int NL = B_ * L_, NS = B_ * S_;
    for (int i = blockIdx.x * blockDim.x + threadIdx.x; i < NL;
         i += gridDim.x * blockDim.x) {
        hlf[i] = 0.0f; hlh[i] = (_Float16)0.0f; clf[i] = 0.0f;
        if (i < NS) { hsf[i] = 0.0f; hsh[i] = (_Float16)0.0f; csf[i] = 0.0f; }
    }
}

// ---------------- async B-tile staging (CDNA5 async global->LDS path) ----------
// Each of the 128 threads copies the 32-byte B fragment for (gate = tid>>5,
// lane = tid&31) of the current 32-K slice into LDS, pre-swizzled into the
// per-lane WMMA fragment layout. Tracked by ASYNCcnt; overlapped with WMMAs.
__device__ __forceinline__ void stage_b_async(const _Float16* __restrict__ W,
                                              int H, int K, int d0, int kb,
                                              const _Float16* ldsDst, int tid) {
    const int g = tid >> 5;
    const int n = d0 + (tid & 15);
    const int khalf = (tid >> 4) & 1;
    const _Float16* src = W + (size_t)(g * H + n) * K + kb + 16 * khalf;
    // AMDGCN flat addresses to LDS carry the LDS byte offset in bits [31:0].
    unsigned lds = (unsigned)(uintptr_t)ldsDst + (unsigned)tid * 32u;
    unsigned long long ga = (unsigned long long)(uintptr_t)src;
    asm volatile(
        "global_load_async_to_lds_b128 %0, %2, off\n\t"
        "global_load_async_to_lds_b128 %1, %3, off"
        :
        : "v"(lds), "v"(lds + 16u), "v"(ga), "v"(ga + 16ull)
        : "memory");
}

// ---------------- per-step cell kernel (WMMA) ----------------
// grid = 80 blocks: blocks 0..63 -> large cell (16 hidden units each),
//                   blocks 64..79 -> small cell. 128 threads = 4 waves;
// wave w handles batch rows [16w,16w+16). B tiles (shared by all 4 waves) are
// double-buffered through LDS via async copies; each wave accumulates the 4
// gate tiles (i,f,g,o) for its 16x16 patch and fuses the LSTM pointwise
// update in the epilogue.
__global__ __launch_bounds__(128) void step_cells(
    int t,
    const _Float16* __restrict__ embh,
    const _Float16* __restrict__ wfl, const _Float16* __restrict__ wfs,
    const float* __restrict__ bias_l, const float* __restrict__ bias_s,
    const _Float16* __restrict__ hl_h16_in, const _Float16* __restrict__ hs_h16_in,
    const float* __restrict__ cl_in, const float* __restrict__ cs_in,
    float* __restrict__ hl2_tmp, float* __restrict__ cl2_tmp,
    float* __restrict__ hs_out_f32, _Float16* __restrict__ hs_out_f16,
    float* __restrict__ cs_out) {

    __shared__ __align__(32) _Float16 sB[2][4 * 32 * 16];   // 2 x 4KB, fragment layout

    const int tid  = threadIdx.x;
    const int lane = tid & 31;
    const int wave = tid >> 5;
    const bool large = (blockIdx.x < (L_ / 16));
    const int d0 = large ? blockIdx.x * 16 : (blockIdx.x - L_ / 16) * 16;
    const int H  = large ? L_ : S_;
    const int K  = large ? KL_ : KS_;
    const _Float16* __restrict__ W    = large ? wfl : wfs;
    const float* __restrict__ bias    = large ? bias_l : bias_s;
    const _Float16* __restrict__ hf16 = large ? hl_h16_in : hs_h16_in;
    const float* __restrict__ cin     = large ? cl_in : cs_in;

    const int mrow = wave * 16;       // batch-row tile base
    const int lrow = lane & 15;       // A: row within tile / B,C: column
    const int lhi  = lane >> 4;       // K half selector (A/B), M half (C)

    // init accumulators with fused bias (bias depends only on column)
    v8f acc[4];
#pragma unroll
    for (int g = 0; g < 4; ++g) {
        float bv = bias[g * H + d0 + lrow];
#pragma unroll
        for (int r = 0; r < 8; ++r) acc[g][r] = bv;
    }

    // prime buffer 0
    stage_b_async(W, H, K, d0, 0, &sB[0][0], tid);
    asm volatile("s_wait_asynccnt 0" ::: "memory");
    __syncthreads();

    const int nkb = K >> 5;
    for (int it = 0; it < nkb; ++it) {
        const int kb = it << 5;
        const int buf = it & 1;
        if (it + 1 < nkb)
            stage_b_async(W, H, K, d0, kb + 32, &sB[buf ^ 1][0], tid);

        // ---- A fragment: 16x32 f16, rows = batch rows ----
        const _Float16* abase =
            (kb < E_) ? (embh + ((size_t)(mrow + lrow) * T_ + t) * E_ + kb)
                      : (hf16 + (size_t)(mrow + lrow) * H + (kb - E_));
        __builtin_prefetch((const void*)(abase + 32), 0, 1);
        v16h a;
#pragma unroll
        for (int v = 0; v < 8; ++v) {
            int ko = 2 * v + 16 * lhi;
            a[2 * v]     = abase[ko];
            a[2 * v + 1] = abase[ko + 1];
        }
        // ---- 4 gate B fragments from LDS + WMMA ----
#pragma unroll
        for (int g = 0; g < 4; ++g) {
            const v16h bf = *(const v16h*)&sB[buf][(g * 32 + lane) * 16];
            acc[g] = __builtin_amdgcn_wmma_f32_16x16x32_f16(
                false, a, false, bf, (short)0, acc[g], false, false);
        }
        // next buffer ready + everyone done reading this buffer
        asm volatile("s_wait_asynccnt 0" ::: "memory");
        __syncthreads();
    }

    // ---- fused LSTM pointwise epilogue ----
    const int n = d0 + lrow;                 // hidden unit index
    const int mbase = mrow + lhi * 8;        // C layout: lanes>=16 hold M=8..15
#pragma unroll
    for (int r = 0; r < 8; ++r) {
        int row = mbase + r;
        float iv = acc[0][r], fv = acc[1][r], gv = acc[2][r], ov = acc[3][r];
        float cprev = cin[(size_t)row * H + n];
        float c2 = sigm_(fv) * cprev + sigm_(iv) * tanhf(gv);
        float h2 = sigm_(ov) * tanhf(c2);
        size_t o = (size_t)row * H + n;
        if (large) {
            hl2_tmp[o] = h2; cl2_tmp[o] = c2;
        } else {  // small cell state updates unconditionally
            hs_out_f32[o] = h2; hs_out_f16[o] = (_Float16)h2; cs_out[o] = c2;
        }
    }
}

// ---------------- per-step gate + blend kernel ----------------
__global__ __launch_bounds__(256) void step_gate(
    int t,
    const int* __restrict__ x, const float* __restrict__ u,
    const float* __restrict__ embedding,
    const float* __restrict__ W_lin, const float* __restrict__ b_lin,
    const float* __restrict__ hl2, const float* __restrict__ cl2,
    const float* __restrict__ hl_in,                     // parity pin (old h_l)
    const float* __restrict__ hs_new, const float* __restrict__ cs_new,  // parity pout
    float* __restrict__ hl_out, _Float16* __restrict__ hl_out_f16,
    float* __restrict__ cl_out,
    float* __restrict__ out_hseq, float* __restrict__ out_pseq) {

    const int b = blockIdx.x, tid = threadIdx.x;
    __shared__ float s0[256], s1[256], sc[4];

    float z0 = 0.0f, z1 = 0.0f;
    const size_t eb = (size_t)x[(size_t)b * T_ + t] * E_;
    for (int k = tid; k < E_ + 2 * L_; k += 256) {
        float v;
        if (k < E_)            v = embedding[eb + k];
        else if (k < E_ + L_)  v = hl2[(size_t)b * L_ + (k - E_)];
        else                   v = cl2[(size_t)b * L_ + (k - E_ - L_)];
        z0 += v * W_lin[k];
        z1 += v * W_lin[(E_ + 2 * L_) + k];
    }
    s0[tid] = z0; s1[tid] = z1;
    __syncthreads();
    for (int off = 128; off > 0; off >>= 1) {
        if (tid < off) { s0[tid] += s0[tid + off]; s1[tid] += s1[tid + off]; }
        __syncthreads();
    }
    if (tid == 0) {
        float a0 = s0[0] + b_lin[0], a1 = s1[0] + b_lin[1];
        float m = fmaxf(a0, a1);
        float e0 = __expf(a0 - m), e1 = __expf(a1 - m);
        float se = e0 + e1;
        float p0 = e0 / se, p1 = e1 / se;
        float lse = m + __logf(se);
        float u0 = fminf(fmaxf(u[((size_t)b * T_ + t) * 2 + 0], 1e-9f), 1.0f - 1e-7f);
        float u1 = fminf(fmaxf(u[((size_t)b * T_ + t) * 2 + 1], 1e-9f), 1.0f - 1e-7f);
        float g0 = -__logf(-__logf(u0)), g1 = -__logf(-__logf(u1));
        float y0 = (a0 - lse + g0) / TAU_, y1 = (a1 - lse + g1) / TAU_;
        float mm = fmaxf(y0, y1);
        float r0 = __expf(y0 - mm), r1 = __expf(y1 - mm);
        float sr = r0 + r1;
        sc[0] = p0; sc[1] = p1; sc[2] = r0 / sr; sc[3] = r1 / sr;
    }
    __syncthreads();
    const float r0 = sc[2], r1 = sc[3];

    for (int d = tid; d < L_; d += 256) {
        size_t o = (size_t)b * L_ + d;
        float h2 = hl2[o], c2 = cl2[o];
        float hc = (d < S_) ? hs_new[(size_t)b * S_ + d] : hl_in[o];
        float cc = (d < S_) ? cs_new[(size_t)b * S_ + d] : c2;
        float hn = r0 * h2 + r1 * hc;
        float cn = r0 * c2 + r1 * cc;
        hl_out[o] = hn; hl_out_f16[o] = (_Float16)hn; cl_out[o] = cn;
        out_hseq[((size_t)b * T_ + t) * L_ + d] = hn;
    }
    if (tid < 2) out_pseq[((size_t)b * T_ + t) * 2 + tid] = sc[tid];
}

// ---------------- classifier head ----------------
__global__ __launch_bounds__(256) void classifier(
    const float* __restrict__ hfin,
    const float* __restrict__ W1, const float* __restrict__ b1,
    const float* __restrict__ W2, const float* __restrict__ b2,
    float* __restrict__ out_logits) {
    const int b = blockIdx.x, tid = threadIdx.x;
    __shared__ float sh[L_];
    __shared__ float sa[512];
    __shared__ float sl[NC_];
    for (int d = tid; d < L_; d += 256) sh[d] = selu_(hfin[(size_t)b * L_ + d]);
    __syncthreads();
    for (int j = tid; j < 512; j += 256) {
        float acc = b1[j];
        const float* wr = W1 + (size_t)j * L_;
        for (int k = 0; k < L_; ++k) acc += sh[k] * wr[k];
        sa[j] = selu_(acc);
    }
    __syncthreads();
    if (tid < NC_) {
        float acc = b2[tid];
        const float* wr = W2 + (size_t)tid * 512;
        for (int k = 0; k < 512; ++k) acc += sa[k] * wr[k];
        sl[tid] = acc;
    }
    __syncthreads();
    if (tid == 0) {
        float m = sl[0];
        for (int j = 1; j < NC_; ++j) m = fmaxf(m, sl[j]);
        float se = 0.0f, e[NC_];
        for (int j = 0; j < NC_; ++j) { e[j] = __expf(sl[j] - m); se += e[j]; }
        for (int j = 0; j < NC_; ++j) out_logits[(size_t)b * NC_ + j] = e[j] / se;
    }
}

// ---------------- launch ----------------
extern "C" void kernel_launch(void* const* d_in, const int* in_sizes, int n_in,
                              void* d_out, int out_size, void* d_ws, size_t ws_size,
                              hipStream_t stream) {
    (void)in_sizes; (void)n_in; (void)out_size; (void)ws_size;
    const int*   x     = (const int*)d_in[0];
    const float* u     = (const float*)d_in[1];
    const float* emb   = (const float*)d_in[2];
    const float* Wihl  = (const float*)d_in[3];
    const float* Whhl  = (const float*)d_in[4];
    const float* bihl  = (const float*)d_in[5];
    const float* bhhl  = (const float*)d_in[6];
    const float* Wihs  = (const float*)d_in[7];
    const float* Whhs  = (const float*)d_in[8];
    const float* bihs  = (const float*)d_in[9];
    const float* bhhs  = (const float*)d_in[10];
    const float* W_lin = (const float*)d_in[11];
    const float* b_lin = (const float*)d_in[12];
    const float* W1    = (const float*)d_in[13];
    const float* b1    = (const float*)d_in[14];
    const float* W2    = (const float*)d_in[15];
    const float* b2    = (const float*)d_in[16];

    // ---- carve workspace ----
    char* ws = (char*)d_ws;
    size_t off = 0;
    auto take = [&](size_t bytes) -> char* {
        char* p = ws + off;
        off = (off + bytes + 255) & ~(size_t)255;
        return p;
    };
    _Float16* wfl    = (_Float16*)take((size_t)4 * L_ * KL_ * 2);
    _Float16* wfs    = (_Float16*)take((size_t)4 * S_ * KS_ * 2);
    float*    bias_l = (float*)take((size_t)4 * L_ * 4);
    float*    bias_s = (float*)take((size_t)4 * S_ * 4);
    _Float16* embh   = (_Float16*)take((size_t)B_ * T_ * E_ * 2);
    float*    hlf[2]; _Float16* hlh[2]; float* clf[2];
    float*    hsf[2]; _Float16* hsh[2]; float* csf[2];
    for (int p = 0; p < 2; ++p) {
        hlf[p] = (float*)take((size_t)B_ * L_ * 4);
        hlh[p] = (_Float16*)take((size_t)B_ * L_ * 2);
        clf[p] = (float*)take((size_t)B_ * L_ * 4);
        hsf[p] = (float*)take((size_t)B_ * S_ * 4);
        hsh[p] = (_Float16*)take((size_t)B_ * S_ * 2);
        csf[p] = (float*)take((size_t)B_ * S_ * 4);
    }
    float* hl2 = (float*)take((size_t)B_ * L_ * 4);
    float* cl2 = (float*)take((size_t)B_ * L_ * 4);

    float* out_logits = (float*)d_out;
    float* out_hseq   = out_logits + (size_t)B_ * NC_;
    float* out_pseq   = out_hseq + (size_t)B_ * T_ * L_;

    // ---- prep (runs every call; deterministic) ----
    prep_weights<<<4096, 256, 0, stream>>>(Wihl, Whhl, bihl, bhhl,
                                           Wihs, Whhs, bihs, bhhs,
                                           wfl, wfs, bias_l, bias_s);
    prep_embed<<<8192, 256, 0, stream>>>(x, emb, embh);
    prep_zero<<<256, 256, 0, stream>>>(hlf[0], hlh[0], clf[0], hsf[0], hsh[0], csf[0]);

    // ---- sequential scan ----
    for (int t = 0; t < T_; ++t) {
        int pin = t & 1, pout = pin ^ 1;
        step_cells<<<L_ / 16 + S_ / 16, 128, 0, stream>>>(
            t, embh, wfl, wfs, bias_l, bias_s,
            hlh[pin], hsh[pin], clf[pin], csf[pin],
            hl2, cl2, hsf[pout], hsh[pout], csf[pout]);
        step_gate<<<B_, 256, 0, stream>>>(
            t, x, u, emb, W_lin, b_lin,
            hl2, cl2, hlf[pin], hsf[pout], csf[pout],
            hlf[pout], hlh[pout], clf[pout], out_hseq, out_pseq);
    }

    // final state lives in parity 0 after t=511 (pout = 0)
    classifier<<<B_, 256, 0, stream>>>(hlf[0], W1, b1, W2, b2, out_logits);
}
